// UserUserAggregator_73461120631291
// MI455X (gfx1250) — matrix operand order repaired and checked by reading, not compiled, not run on hardware
//
#include <hip/hip_runtime.h>
#include <hip/hip_bf16.h>

typedef __attribute__((ext_vector_type(16))) _Float16 v16h;
typedef __attribute__((ext_vector_type(8)))  _Float16 v8h;
typedef __attribute__((ext_vector_type(8)))  float    v8f;
typedef __attribute__((ext_vector_type(4)))  float    v4f;

#define DIMS 128
#define KNB  32
#define LROW 136   // LDS row stride in halfs (272B = 68 dwords -> staggered banks)

static __device__ __forceinline__ v8f wmma16x16x32(v16h a, v16h b, v8f c) {
  // D = A(16x32 f16) x B(32x16 f16) + C(16x16 f32)
  return __builtin_amdgcn_wmma_f32_16x16x32_f16(false, a, false, b, (short)0, c, false, false);
}

// ---------------------------------------------------------------------------
// Pack w1 [256x128] and w2 [128x128] (f32, row-major [in,out]) into WMMA
// B-fragment order (f16): frag(ks,nt) -> 32 lanes x 16 halfs contiguous.
// Lane l holds column n = nt*16 + (l&15), K = ks*32 + (l>=16?8:0) + (j<8 ? j : j+8), j=0..15.
// ---------------------------------------------------------------------------
__global__ void pack_weights_kernel(const float* __restrict__ w1,
                                    const float* __restrict__ w2,
                                    _Float16* __restrict__ ws) {
  const int total1 = 64 * 512;   // 8 ks * 8 nt fragments of w1
  const int total2 = 32 * 512;   // 4 ks * 8 nt fragments of w2
  int tid = blockIdx.x * blockDim.x + threadIdx.x;
  int nthreads = gridDim.x * blockDim.x;
  for (int idx = tid; idx < total1 + total2; idx += nthreads) {
    int i = idx;
    const float* w;
    _Float16* dst;
    if (i < total1) { w = w1; dst = ws; }
    else            { i -= total1; w = w2; dst = ws + total1; }
    int frag   = i >> 9;       // /512
    int within = i & 511;
    int lane   = within >> 4;
    int j      = within & 15;
    int ks     = frag >> 3;
    int nt     = frag & 7;
    int k = ks * 32 + ((lane >= 16) ? 8 : 0) + ((j < 8) ? j : j + 8);
    int n = nt * 16 + (lane & 15);
    dst[(frag << 9) + within] = (_Float16)w[k * DIMS + n];
  }
}

// ---------------------------------------------------------------------------
// One wave (32 lanes) per node. 4 waves / block.
// ---------------------------------------------------------------------------
__global__ void __launch_bounds__(128)
user_agg_kernel(const int*   __restrict__ nodes,
                const int*   __restrict__ neighbours,
                const float* __restrict__ user_emb,
                const float* __restrict__ b1,
                const float* __restrict__ b2,
                const float* __restrict__ w3,
                const float* __restrict__ b3,
                const _Float16* __restrict__ wfrag,
                float* __restrict__ out,
                int n_nodes) {
  __shared__ _Float16 lds[4 * 32 * LROW];

  const int lane  = threadIdx.x & 31;
  const int wave  = threadIdx.x >> 5;
  const int node0 = blockIdx.x * 4 + wave;
  const int node  = (node0 < n_nodes) ? node0 : (n_nodes - 1);

  _Float16* myL = lds + wave * (32 * LROW);
  const int lh  = (lane >= 16) ? 1 : 0;   // lane half
  const int lo  = lh ? 8 : 0;             // A-layout K sub-offset
  const int l15 = lane & 15;

  // Per-M-tile neighbor row pointers (lane = row within tile)
  const float* rowp[2];
  #pragma unroll
  for (int mt = 0; mt < 2; ++mt) {
    int nb = neighbours[(size_t)node * KNB + mt * 16 + l15];
    rowp[mt] = user_emb + (size_t)nb * DIMS;
  }
  const float* prow = user_emb + (size_t)nodes[node] * DIMS;

  const v16h* bf1 = (const v16h*)wfrag;              // 64 frags (layer 1)
  const v16h* bf2 = (const v16h*)(wfrag + 64 * 512); // 32 frags (layer 2)

  // ======================= Layer 1: [32x256]@[256x128] =====================
  v8f acc[2][8] = {};
  {
    v4f araw[2][4];
    // prologue: issue A gathers for ks=0 and the first B fragment
    #pragma unroll
    for (int mt = 0; mt < 2; ++mt) {
      const float* src = rowp[mt];
      araw[mt][0] = *(const v4f*)(src + lo);
      araw[mt][1] = *(const v4f*)(src + lo + 4);
      araw[mt][2] = *(const v4f*)(src + 16 + lo);
      araw[mt][3] = *(const v4f*)(src + 16 + lo + 4);
    }
    v16h bcur = bf1[lane];

    #pragma unroll
    for (int ks = 0; ks < 8; ++ks) {
      // convert the raw f32 chunks fetched one iteration ago
      v16h a[2];
      #pragma unroll
      for (int mt = 0; mt < 2; ++mt)
        #pragma unroll
        for (int c = 0; c < 4; ++c)
          #pragma unroll
          for (int i = 0; i < 4; ++i)
            a[mt][c * 4 + i] = (_Float16)araw[mt][c][i];

      // issue next ks's A gathers early (covers L2/HBM latency with WMMAs)
      if (ks < 7) {
        int ksn = ks + 1;
        #pragma unroll
        for (int mt = 0; mt < 2; ++mt) {
          const float* src = (ksn < 4) ? (rowp[mt] + ksn * 32)
                                       : (prow + (ksn - 4) * 32);
          araw[mt][0] = *(const v4f*)(src + lo);
          araw[mt][1] = *(const v4f*)(src + lo + 4);
          araw[mt][2] = *(const v4f*)(src + 16 + lo);
          araw[mt][3] = *(const v4f*)(src + 16 + lo + 4);
        }
      }

      // B double-buffer: fetch frag f+1 before consuming frag f
      #pragma unroll
      for (int nt = 0; nt < 8; ++nt) {
        int f = ks * 8 + nt;
        v16h bnext = bcur;
        if (f < 63) bnext = bf1[(f + 1) * 32 + lane];
        acc[0][nt] = wmma16x16x32(a[0], bcur, acc[0][nt]);
        acc[1][nt] = wmma16x16x32(a[1], bcur, acc[1][nt]);
        bcur = bnext;
      }
    }
  }

  // bias + ReLU, stage h1 (f16) to LDS in row-major for A-layout reload
  #pragma unroll
  for (int mt = 0; mt < 2; ++mt) {
    #pragma unroll
    for (int nt = 0; nt < 8; ++nt) {
      int col = nt * 16 + l15;
      float bias = b1[col];
      #pragma unroll
      for (int v = 0; v < 8; ++v) {
        int row = mt * 16 + lh * 8 + v;            // C layout: lanes16-31 -> M+8
        float h = acc[mt][nt][v] + bias;
        myL[row * LROW + col] = (_Float16)fmaxf(h, 0.0f);
      }
    }
  }

  // ======================= Layer 2: [32x128]@[128x128] =====================
  v8f acc2[2][8] = {};
  {
    v8h p[2][2];
    #pragma unroll
    for (int mt = 0; mt < 2; ++mt) {
      const _Float16* s = myL + (mt * 16 + l15) * LROW + lo;
      p[mt][0] = *(const v8h*)s;
      p[mt][1] = *(const v8h*)(s + 16);
    }
    v16h bcur = bf2[lane];

    #pragma unroll
    for (int ks = 0; ks < 4; ++ks) {
      v16h a[2];
      #pragma unroll
      for (int mt = 0; mt < 2; ++mt)
        #pragma unroll
        for (int i = 0; i < 8; ++i) {
          a[mt][i]     = p[mt][0][i];
          a[mt][8 + i] = p[mt][1][i];
        }

      if (ks < 3) {
        int ksn = ks + 1;
        #pragma unroll
        for (int mt = 0; mt < 2; ++mt) {
          const _Float16* s = myL + (mt * 16 + l15) * LROW + ksn * 32 + lo;
          p[mt][0] = *(const v8h*)s;
          p[mt][1] = *(const v8h*)(s + 16);
        }
      }

      #pragma unroll
      for (int nt = 0; nt < 8; ++nt) {
        int f = ks * 8 + nt;
        v16h bnext = bcur;
        if (f < 31) bnext = bf2[(f + 1) * 32 + lane];
        acc2[0][nt] = wmma16x16x32(a[0], bcur, acc2[0][nt]);
        acc2[1][nt] = wmma16x16x32(a[1], bcur, acc2[1][nt]);
        bcur = bnext;
      }
    }
  }

  // bias + ReLU, stage h2 (f16) to LDS (overwrites h1)
  #pragma unroll
  for (int mt = 0; mt < 2; ++mt) {
    #pragma unroll
    for (int nt = 0; nt < 8; ++nt) {
      int col = nt * 16 + l15;
      float bias = b2[col];
      #pragma unroll
      for (int v = 0; v < 8; ++v) {
        int row = mt * 16 + lh * 8 + v;
        float h = acc2[mt][nt][v] + bias;
        myL[row * LROW + col] = (_Float16)fmaxf(h, 0.0f);
      }
    }
  }

  // ================= scores + softmax over the 32 neighbors ================
  // lane k computes score_k = h2[k,:] . w3 + b3
  float s = b3[0];
  {
    const _Float16* hr = myL + lane * LROW;
    #pragma unroll
    for (int j = 0; j < DIMS; j += 8) {
      v8h hh = *(const v8h*)(hr + j);
      v4f wa = *(const v4f*)(w3 + j);
      v4f wb = *(const v4f*)(w3 + j + 4);
      #pragma unroll
      for (int i = 0; i < 4; ++i) {
        s += (float)hh[i]     * wa[i];
        s += (float)hh[4 + i] * wb[i];
      }
    }
  }
  float m = s;
  #pragma unroll
  for (int off = 16; off > 0; off >>= 1) m = fmaxf(m, __shfl_xor(m, off, 32));
  float e = __expf(s - m);
  float sum = e;
  #pragma unroll
  for (int off = 16; off > 0; off >>= 1) sum += __shfl_xor(sum, off, 32);
  float attn = e / sum;

  // ================= weighted sum: out[n,:] = attn^T @ neigh ===============
  // lane handles 4 output dims; neighbor rows are L2-resident re-reads
  float o0 = 0.f, o1 = 0.f, o2 = 0.f, o3 = 0.f;
  const int d0 = lane * 4;
  #pragma unroll 4
  for (int k = 0; k < KNB; ++k) {
    float a = __shfl(attn, k, 32);
    int nb = neighbours[(size_t)node * KNB + k];     // wave-uniform -> scalar
    v4f x = *(const v4f*)(user_emb + (size_t)nb * DIMS + d0);
    o0 += a * x[0]; o1 += a * x[1]; o2 += a * x[2]; o3 += a * x[3];
  }
  if (node0 < n_nodes) {
    v4f o = {o0, o1, o2, o3};
    *(v4f*)(out + (size_t)node * DIMS + d0) = o;
  }
}

// ---------------------------------------------------------------------------
extern "C" void kernel_launch(void* const* d_in, const int* in_sizes, int n_in,
                              void* d_out, int out_size, void* d_ws, size_t ws_size,
                              hipStream_t stream) {
  const int*   nodes      = (const int*)d_in[0];
  const int*   neighbours = (const int*)d_in[1];
  const float* user_emb   = (const float*)d_in[2];
  const float* w1 = (const float*)d_in[3];
  const float* b1 = (const float*)d_in[4];
  const float* w2 = (const float*)d_in[5];
  const float* b2 = (const float*)d_in[6];
  const float* w3 = (const float*)d_in[7];
  const float* b3 = (const float*)d_in[8];
  float* out = (float*)d_out;
  const int n_nodes = in_sizes[0];

  _Float16* wfrag = (_Float16*)d_ws;   // needs 96 KB: (64+32) frags * 512 halfs

  pack_weights_kernel<<<64, 256, 0, stream>>>(w1, w2, wfrag);

  int blocks = (n_nodes + 3) / 4;      // 4 waves/block, one node per wave
  user_agg_kernel<<<blocks, 128, 0, stream>>>(nodes, neighbours, user_emb,
                                              b1, b2, w3, b3, wfrag, out, n_nodes);
}
